// DPXTokenizer_50629074485721
// MI455X (gfx1250) — matrix operand order repaired and checked by reading, not compiled, not run on hardware
//
#include <hip/hip_runtime.h>
#include <hip/hip_bf16.h>

typedef __attribute__((ext_vector_type(16))) _Float16 v16h;
typedef __attribute__((ext_vector_type(8)))  float    v8f;
typedef __attribute__((ext_vector_type(4)))  unsigned u32x4;
typedef __attribute__((ext_vector_type(8)))  unsigned u32x8;

#define Bn   8
#define Cc   3
#define Hh   256
#define Ww   256
#define HIDF 32
#define Ntot (Bn*Hh*Ww)                 // 524288
#define EH   (Bn*Hh*(Ww-1))            // 522240
#define EV   (Bn*(Hh-1)*Ww)            // 522240
#define ETOT (2*EH + 2*EV)             // 2088960
#define CMPT 0.1f

// ---------------- ordered-float encoding for atomic max ----------------
__device__ __forceinline__ unsigned encf(float f) {
  unsigned u = __float_as_uint(f);
  return (u & 0x80000000u) ? ~u : (u | 0x80000000u);
}
__device__ __forceinline__ float decf(unsigned k) {
  return (k & 0x80000000u) ? __uint_as_float(k & 0x7FFFFFFFu)
                           : __uint_as_float(~k);
}

// ---------------- edge enumeration (matches la/lb concat order) --------
__device__ __forceinline__ void edge_nodes(int e, int& a, int& b) {
  bool sw = false;
  if (e < 2*EH) {
    if (e >= EH) { e -= EH; sw = true; }
    int im = e / (Hh*(Ww-1));
    int r  = e - im*(Hh*(Ww-1));
    int y  = r / (Ww-1);
    int x  = r - y*(Ww-1);
    a = (im*Hh + y)*Ww + x;  b = a + 1;
  } else {
    e -= 2*EH;
    if (e >= EV) { e -= EV; sw = true; }
    int im = e / ((Hh-1)*Ww);
    int r  = e - im*((Hh-1)*Ww);
    int y  = r / Ww;
    int x  = r - y*Ww;
    a = (im*Hh + y)*Ww + x;  b = a + Ww;
  }
  if (sw) { int t = a; a = b; b = t; }
}

// ======================================================================
// Conv 3x3 (C=3 -> HID=32) as im2col GEMM on v_wmma_f32_16x16x32_f16.
// - Weights DMA'd into LDS by the Tensor Data Mover (tensor_load_to_lds),
//   synced with s_wait_tensorcnt, then converted to f16 in LDS.
// - Input strip (3 chan x 3 rows x 130 cols) staged in LDS with coalesced
//   loads; B fragments come from ds_load, not divergent global gathers.
// One wave: 16 pixels x 32 out-channels (two M-tiles of 16), K=27 pad 32.
// Block: 256 threads = 8 waves = 128 contiguous pixels of one image row.
// ======================================================================
__global__ void dpx_conv_wmma(const float* __restrict__ img,
                              const float* __restrict__ cw,
                              const float* __restrict__ cb,
                              float* __restrict__ fV) {
  __shared__ float    wf32[32*27];     // raw f32 weights, filled by TDM
  __shared__ _Float16 wlds[32][32];    // f16 weights, K padded 27->32
  __shared__ _Float16 tile[9][132];    // [c*3+ky][col] input strip, 130 used

  const int tid = threadIdx.x;
  const int nb  = blockIdx.x * 128;    // first pixel of block
  const int b   = nb >> 16;            // / (Hh*Ww)
  const int rem = nb & 0xFFFF;
  const int y   = rem >> 8;            // / Ww
  const int xb  = rem & 0xFF;          // 0 or 128

  // ---- 1) TDM: DMA the 32x27 f32 weight tensor into LDS (wave 0 only) ----
  if (tid < 32) {
    unsigned long long ga = (unsigned long long)(uintptr_t)cw;
    unsigned ldsa = (unsigned)(uintptr_t)(&wf32[0]);
    u32x4 g0;
    g0[0] = 1u;                                   // count=1, user D#
    g0[1] = ldsa;                                 // lds_addr (bytes)
    g0[2] = (unsigned)ga;                         // global_addr[31:0]
    g0[3] = ((unsigned)(ga >> 32) & 0x01FFFFFFu)  // global_addr[56:32]
            | (2u << 30);                         // type=2 ("image")
    u32x8 g1;
    g1[0] = 2u << 16;          // workgroup_mask=0, data_size=2 (4 bytes)
    g1[1] = 27u << 16;         // tensor_dim0 = 27 (low 16 bits)
    g1[2] = 32u << 16;         // tensor_dim0 hi=0 | tensor_dim1 = 32
    g1[3] = 27u << 16;         // tensor_dim1 hi=0 | tile_dim0 = 27
    g1[4] = 32u;               // tile_dim1 = 32, tile_dim2 = 0
    g1[5] = 27u;               // tensor_dim0_stride = 27
    g1[6] = 0u;
    g1[7] = 0u;
    asm volatile("tensor_load_to_lds %0, %1" :: "s"(g0), "s"(g1) : "memory");
    __builtin_amdgcn_s_wait_tensorcnt(0);
  }

  // ---- 2) coalesced stage of input strip (cols xb-1 .. xb+128) ----
  for (int i = tid; i < 9*130; i += 256) {
    int r9  = i / 130;                 // c*3 + ky
    int col = i - r9*130;
    int c   = r9 / 3;
    int ky  = r9 - c*3;
    int yy  = y + ky - 1;
    int xx  = xb + col - 1;
    float v = 0.0f;
    if (yy >= 0 && yy < Hh && xx >= 0 && xx < Ww)
      v = img[((size_t)(b*Cc + c)*Hh + yy)*Ww + xx];
    tile[r9][col] = (_Float16)v;
  }
  __syncthreads();                      // TDM data + tile stores visible

  // ---- 3) convert TDM'd weights f32 -> f16 with K padding ----
  for (int i = tid; i < 32*32; i += 256) {
    int oc = i >> 5, k = i & 31;
    wlds[oc][k] = (k < 27) ? (_Float16)wf32[oc*27 + k] : (_Float16)0.0f;
  }
  __syncthreads();

  // ---- 4) WMMA fragments ----
  const int wave = tid >> 5;
  const int lane = tid & 31;
  const int hgrp = lane >> 4;           // kbase 0/8 (A and B)
  const int l15  = lane & 15;
  const int px   = wave*16 + l15;       // pixel offset in block (0..127)
  const int n    = nb + px;

  // B fragment (32x16, K x pixels), from LDS strip
  v16h bf;
  #pragma unroll
  for (int h = 0; h < 16; ++h) {
    int k  = (h < 8) ? h + 8*hgrp : 16 + 8*hgrp + (h - 8);   // 0..31
    int r9 = k / 3;                     // c*3+ky (<=10; clamp for pad lanes)
    int dx = k - r9*3;
    _Float16 vv = tile[(r9 < 9) ? r9 : 0][px + dx];
    bf[h] = (k < 27) ? vv : (_Float16)0.0f;
  }

  #pragma unroll
  for (int t = 0; t < 2; ++t) {
    // A fragment (16x32): M = l15, K interleave with kbase = 8*hgrp
    v16h af;
    #pragma unroll
    for (int h = 0; h < 16; ++h) {
      int vg = h >> 1, j = h & 1;
      int k = ((vg < 4) ? 2*vg : 16 + 2*(vg - 4)) + 8*hgrp + j;
      af[h] = wlds[t*16 + l15][k];
    }
    v8f acc = {};
    acc = __builtin_amdgcn_wmma_f32_16x16x32_f16(false, af, false, bf,
                                                 (short)0, acc, false, false);
    #pragma unroll
    for (int r = 0; r < 8; ++r) {
      int oc = t*16 + r + 8*hgrp;       // D: M = r + 8*half
      fV[(size_t)n*HIDF + oc] = acc[r] + cb[oc];
    }
  }
}

// ======================================================================
// Region-merge machinery
// ======================================================================
__global__ void dpx_fill_f32(float* __restrict__ p, float v, long n) {
  long i = (long)blockIdx.x*256 + threadIdx.x;
  if (i < n) p[i] = v;
}
__global__ void dpx_iota(int* __restrict__ p) {
  int i = blockIdx.x*256 + threadIdx.x;
  if (i < Ntot) p[i] = i;
}
__global__ void dpx_init_ms_best(unsigned* __restrict__ ms, int* __restrict__ best) {
  int i = blockIdx.x*256 + threadIdx.x;
  if (i < Ntot) { ms[i] = 0u; best[i] = -1; }
}
__global__ void dpx_seg_accum(const float* __restrict__ fV,
                              const int* __restrict__ labels,
                              float* __restrict__ seg, float* __restrict__ cnt) {
  long t = (long)blockIdx.x*256 + threadIdx.x;
  if (t >= (long)Ntot*HIDF) return;
  int n = (int)(t >> 5), d = (int)(t & 31);
  int l = labels[n];
  atomicAdd(&seg[(size_t)l*HIDF + d], fV[t]);
  if (d == 0) atomicAdd(&cnt[l], 1.0f);
}
__global__ void dpx_seg_mean(float* __restrict__ seg, const float* __restrict__ cnt) {
  long t = (long)blockIdx.x*256 + threadIdx.x;
  if (t >= (long)Ntot*HIDF) return;
  seg[t] = seg[t] / fmaxf(cnt[t >> 5], 1.0f);
}
__global__ void dpx_edge_sim(const int* __restrict__ labels,
                             const float* __restrict__ f,
                             unsigned* __restrict__ msim, float* __restrict__ sim) {
  int e = blockIdx.x*256 + threadIdx.x;
  if (e >= ETOT) return;
  int a, b; edge_nodes(e, a, b);
  int ra = labels[a], rb = labels[b];
  float s;
  if (ra != rb) {
    const float* fa = f + (size_t)ra*HIDF;
    const float* fb = f + (size_t)rb*HIDF;
    float d2 = 0.0f;
    #pragma unroll
    for (int d = 0; d < HIDF; ++d) { float t = fa[d] - fb[d]; d2 += t*t; }
    s = __expf(-d2);
  } else {
    s = -1.0f;
  }
  sim[e] = s;
  atomicMax(&msim[ra], encf(s));
}
__global__ void dpx_edge_best(const int* __restrict__ labels,
                              const float* __restrict__ sim,
                              const unsigned* __restrict__ msim,
                              int* __restrict__ best) {
  int e = blockIdx.x*256 + threadIdx.x;
  if (e >= ETOT) return;
  int a, b; edge_nodes(e, a, b);
  int ra = labels[a], rb = labels[b];
  float s = sim[e];
  int cand = (s >= decf(msim[ra])) ? rb : -1;
  atomicMax(&best[ra], cand);
}
__global__ void dpx_make_parent(const unsigned* __restrict__ msim,
                                const int* __restrict__ best, int* __restrict__ par) {
  int i = blockIdx.x*256 + threadIdx.x;
  if (i >= Ntot) return;
  float ms = decf(msim[i]);
  int bst = best[i];
  par[i] = (ms > CMPT && bst >= 0) ? bst : i;
}
__global__ void dpx_cycle_break(const int* __restrict__ pin, int* __restrict__ pout) {
  int i = blockIdx.x*256 + threadIdx.x;
  if (i >= Ntot) return;
  int p = pin[i], pp = pin[p];
  pout[i] = (pp == i && i < p) ? i : p;
}
__global__ void dpx_jump(const int* __restrict__ pin, int* __restrict__ pout) {
  int i = blockIdx.x*256 + threadIdx.x;
  if (i >= Ntot) return;
  pout[i] = pin[pin[i]];
}
__global__ void dpx_relabel(const int* __restrict__ lin, const int* __restrict__ par,
                            int* __restrict__ lout) {
  int i = blockIdx.x*256 + threadIdx.x;
  if (i >= Ntot) return;
  lout[i] = par[lin[i]];
}

// ======================================================================
// Finalization
// ======================================================================
__global__ void dpx_region_feat(const float* __restrict__ fmean,
                                const float* __restrict__ lw,
                                const float* __restrict__ lb,
                                float* __restrict__ rf) {
  int i = blockIdx.x*256 + threadIdx.x;
  if (i >= Ntot) return;
  const float* fm = fmean + (size_t)i*HIDF;
  #pragma unroll
  for (int c = 0; c < Cc; ++c) {
    float s = lb[c];
    #pragma unroll
    for (int d = 0; d < HIDF; ++d) s += fm[d] * lw[c*HIDF + d];
    rf[(size_t)i*Cc + c] = s;
  }
}
__global__ void dpx_pix_accum(const float* __restrict__ img,
                              const int* __restrict__ labels,
                              float* __restrict__ psum, float* __restrict__ pcnt) {
  int n = blockIdx.x*256 + threadIdx.x;
  if (n >= Ntot) return;
  int l = labels[n];
  int b = n >> 16; int rem = n & 0xFFFF;
  #pragma unroll
  for (int c = 0; c < Cc; ++c)
    atomicAdd(&psum[(size_t)l*Cc + c], img[(size_t)(b*Cc + c)*Hh*Ww + rem]);
  atomicAdd(&pcnt[l], 1.0f);
}
__global__ void dpx_finalize(const float* __restrict__ img,
                             const int* __restrict__ labels,
                             const float* __restrict__ rf,
                             const float* __restrict__ psum,
                             const float* __restrict__ pcnt,
                             float* __restrict__ out, int* __restrict__ outLab) {
  int n = blockIdx.x*256 + threadIdx.x;
  if (n >= Ntot) return;
  int l = labels[n];
  float c = fmaxf(pcnt[l], 1.0f);
  int b = n >> 16; int rem = n & 0xFFFF;
  #pragma unroll
  for (int ch = 0; ch < Cc; ++ch) {
    float fp = img[(size_t)(b*Cc + ch)*Hh*Ww + rem];
    out[(size_t)n*Cc + ch] = fp + rf[(size_t)l*Cc + ch] - psum[(size_t)l*Cc + ch] / c;
  }
  outLab[n] = l;
}

static inline unsigned grd(long n) { return (unsigned)((n + 255) / 256); }

extern "C" void kernel_launch(void* const* d_in, const int* in_sizes, int n_in,
                              void* d_out, int out_size, void* d_ws, size_t ws_size,
                              hipStream_t stream) {
  const float* img    = (const float*)d_in[0];
  const float* conv_w = (const float*)d_in[1];
  const float* conv_b = (const float*)d_in[2];
  const float* lin_w  = (const float*)d_in[3];
  const float* lin_b  = (const float*)d_in[4];
  float* out      = (float*)d_out;
  int*   outLab   = (int*)d_out + (size_t)Ntot*Cc;

  char* w = (char*)d_ws;
  float*    fV   = (float*)w;    w += (size_t)Ntot*HIDF*4;
  float*    seg  = (float*)w;    w += (size_t)Ntot*HIDF*4;   // seg then cnt contiguous
  float*    cnt  = (float*)w;    w += (size_t)Ntot*4;
  unsigned* msim = (unsigned*)w; w += (size_t)Ntot*4;
  int*      best = (int*)w;      w += (size_t)Ntot*4;
  int*      pA   = (int*)w;      w += (size_t)Ntot*4;
  int*      pB   = (int*)w;      w += (size_t)Ntot*4;
  int*      lA   = (int*)w;      w += (size_t)Ntot*4;
  int*      lB   = (int*)w;      w += (size_t)Ntot*4;
  float*    sim  = (float*)w;    w += (size_t)ETOT*4;
  float*    psum = (float*)w;    w += (size_t)Ntot*Cc*4;
  float*    rf   = (float*)w;

  // 1) conv backbone: TDM weight DMA + LDS-staged im2col + WMMA
  dpx_conv_wmma<<<Ntot/128, 256, 0, stream>>>(img, conv_w, conv_b, fV);
  dpx_iota<<<grd(Ntot), 256, 0, stream>>>(lA);

  int* labels = lA; int* labelsAlt = lB;
  for (int it = 0; it < 8; ++it) {
    dpx_fill_f32<<<grd((long)Ntot*(HIDF+1)), 256, 0, stream>>>(seg, 0.0f, (long)Ntot*(HIDF+1));
    dpx_seg_accum<<<grd((long)Ntot*HIDF), 256, 0, stream>>>(fV, labels, seg, cnt);
    dpx_seg_mean<<<grd((long)Ntot*HIDF), 256, 0, stream>>>(seg, cnt);
    dpx_init_ms_best<<<grd(Ntot), 256, 0, stream>>>(msim, best);
    dpx_edge_sim<<<grd(ETOT), 256, 0, stream>>>(labels, seg, msim, sim);
    dpx_edge_best<<<grd(ETOT), 256, 0, stream>>>(labels, sim, msim, best);
    dpx_make_parent<<<grd(Ntot), 256, 0, stream>>>(msim, best, pA);
    dpx_cycle_break<<<grd(Ntot), 256, 0, stream>>>(pA, pB);
    int* pin = pB; int* pout = pA;
    for (int j = 0; j < 20; ++j) {
      dpx_jump<<<grd(Ntot), 256, 0, stream>>>(pin, pout);
      int* t = pin; pin = pout; pout = t;
    }
    dpx_relabel<<<grd(Ntot), 256, 0, stream>>>(labels, pin, labelsAlt);
    int* t = labels; labels = labelsAlt; labelsAlt = t;
  }

  // 2) finalize: region means -> linear -> mean injection
  dpx_fill_f32<<<grd((long)Ntot*(HIDF+1)), 256, 0, stream>>>(seg, 0.0f, (long)Ntot*(HIDF+1));
  dpx_seg_accum<<<grd((long)Ntot*HIDF), 256, 0, stream>>>(fV, labels, seg, cnt);
  dpx_seg_mean<<<grd((long)Ntot*HIDF), 256, 0, stream>>>(seg, cnt);
  dpx_region_feat<<<grd(Ntot), 256, 0, stream>>>(seg, lin_w, lin_b, rf);
  dpx_fill_f32<<<grd((long)Ntot*Cc), 256, 0, stream>>>(psum, 0.0f, (long)Ntot*Cc);
  dpx_fill_f32<<<grd(Ntot), 256, 0, stream>>>(cnt, 0.0f, (long)Ntot);
  dpx_pix_accum<<<grd(Ntot), 256, 0, stream>>>(img, labels, psum, cnt);
  dpx_finalize<<<grd(Ntot), 256, 0, stream>>>(img, labels, rf, psum, cnt, out, outLab);
}
